// HSCR_67396626809127
// MI455X (gfx1250) — compile-verified, matrix operand order checked by reading, not run on hardware
//
#include <hip/hip_runtime.h>
#include <hip/hip_bf16.h>

typedef _Float16 half_t;
typedef __attribute__((ext_vector_type(16))) _Float16 v16h;
typedef __attribute__((ext_vector_type(8)))  float    v8f;
typedef unsigned uint32x4 __attribute__((ext_vector_type(4)));
typedef int      int32x8  __attribute__((ext_vector_type(8)));
typedef int      int32x4  __attribute__((ext_vector_type(4)));

union FragH { v16h v; unsigned u[8]; };

// ---------------- sizes ----------------
// K paddings to multiples of 64 (BK): 266->320, 400->448, 2051->2112
#define K1P 320
#define K2P 448
#define K3P 2112
#define N3P 192   // W3 rows padded to 3 column tiles of 64
#define LD3 160   // out3 leading dim (157 logical)
#define BK  64
#define LDSP 72   // padded LDS row stride in halfs: 64 + 8 (bank-conflict free)

// ---------------- TDM: 2D tile (rows x 64 halfs) global -> LDS, padded rows ---
// pad_interval = 32 DWORDs (=128B = one 64-half row), pad_amount = 4 DWORDs (=16B)
__device__ __forceinline__ void tdm_load_tile64(const half_t* gsrc, unsigned lds_off,
                                                int rows, int stride_elems) {
  unsigned long long ga = (unsigned long long)(size_t)gsrc;
  uint32x4 g0;
  g0[0] = 1u;                                   // count=1, user mode, no gather
  g0[1] = lds_off;                              // LDS byte address
  g0[2] = (unsigned)ga;                         // global_addr[31:0]
  g0[3] = (unsigned)((ga >> 32) & 0x01FFFFFFu)  // global_addr[56:32]
        | (2u << 30);                           // type = 2 (image)
  unsigned w0 = (1u << 16)    // data_size = 1 -> 2 bytes
              | (1u << 20)    // pad_enable
              | (4u << 22)    // pad_interval: 32 DWORDs (128B row)
              | (3u << 25);   // pad_amount: 4 DWORDs (16B)
  unsigned td0 = (unsigned)stride_elems;        // tensor_dim0 (>= tile width 64)
  unsigned td1 = 0x100000u;                     // tensor_dim1: plenty of rows
  unsigned w1 = (td0 & 0xFFFFu) << 16;          // [47:32]=atomic_addr=0, [63:48]=td0.lo
  unsigned w2 = (td0 >> 16) | ((td1 & 0xFFFFu) << 16);
  unsigned w3 = (td1 >> 16) | (64u << 16);      // [127:112] tile_dim0 = 64 elems
  unsigned w4 = (unsigned)rows;                 // [143:128] tile_dim1 ; tile_dim2=0
  unsigned w5 = (unsigned)stride_elems;         // tensor_dim0_stride[31:0]
  unsigned w6 = 0;                              // stride0 hi, stride1 lo
  unsigned w7 = 0;                              // stride1 hi
  int32x8 g1;
  g1[0] = (int)w0; g1[1] = (int)w1; g1[2] = (int)w2; g1[3] = (int)w3;
  g1[4] = (int)w4; g1[5] = (int)w5; g1[6] = (int)w6; g1[7] = (int)w7;
  int32x4 z4 = {0, 0, 0, 0};
  int32x8 z8 = {0, 0, 0, 0, 0, 0, 0, 0};
  // clang-23 / therock-10.0 lane: 6-arg form (extra int32x8 group, zero-filled)
  __builtin_amdgcn_tensor_load_to_lds(g0, g1, z4, z4, z8, 0);
}

// ---------------- pack activations to f16 ----------------
// per row: 320 cols A1 | 448 cols A2 | 64 tail cols of A3 (init_cam + zero pad)
__global__ void pack_activations(const float* __restrict__ x,
                                 const float* __restrict__ init_pose,
                                 const float* __restrict__ init_shape,
                                 const float* __restrict__ init_cam,
                                 half_t* __restrict__ A1, half_t* __restrict__ A2,
                                 half_t* __restrict__ A3, int M) {
  const int per_row = K1P + K2P + 64; // 832
  long total = (long)M * per_row;
  for (long g = (long)blockIdx.x * blockDim.x + threadIdx.x; g < total;
       g += (long)gridDim.x * blockDim.x) {
    int row = (int)(g / per_row);
    int c   = (int)(g % per_row);
    if (c < K1P) {
      float v = 0.f;
      if (c < 256) v = x[(size_t)row * 256 + c];
      else if (c < 266) v = init_shape[(size_t)row * 10 + (c - 256)];
      A1[(size_t)row * K1P + c] = (half_t)v;
    } else if (c < K1P + K2P) {
      int c2 = c - K1P;
      float v = 0.f;
      if (c2 < 256) v = x[(size_t)row * 256 + c2];
      else if (c2 < 400) v = init_pose[(size_t)row * 144 + (c2 - 256)];
      A2[(size_t)row * K2P + c2] = (half_t)v;
    } else {
      int c3 = c - (K1P + K2P); // 0..63 -> A3 cols 2048..2111
      float v = (c3 < 3) ? init_cam[(size_t)row * 3 + c3] : 0.f;
      A3[(size_t)row * K3P + 2048 + c3] = (half_t)v;
    }
  }
}

// ---------------- pack weights to f16 + fused bias ----------------
__global__ void pack_weights(const float* __restrict__ fc1_w,
                             const float* __restrict__ fc2_w,
                             const float* __restrict__ ktd_w,
                             const float* __restrict__ decshape_w,
                             const float* __restrict__ deccam_w,
                             const float* __restrict__ ktd_b,
                             const float* __restrict__ decshape_b,
                             const float* __restrict__ deccam_b,
                             half_t* __restrict__ W1, half_t* __restrict__ W2,
                             half_t* __restrict__ W3, float* __restrict__ bias3) {
  const int N1 = 1024 * K1P;
  const int N2 = 1024 * K2P;
  const int N3 = N3P * K3P;
  const int total = N1 + N2 + N3 + N3P;
  for (int e = blockIdx.x * blockDim.x + threadIdx.x; e < total;
       e += gridDim.x * blockDim.x) {
    if (e < N1) {
      int r = e / K1P, c = e % K1P;
      W1[e] = (c < 266) ? (half_t)fc1_w[(size_t)r * 266 + c] : (half_t)0.f;
    } else if (e < N1 + N2) {
      int e2 = e - N1;
      int r = e2 / K2P, c = e2 % K2P;
      W2[e2] = (c < 400) ? (half_t)fc2_w[(size_t)r * 400 + c] : (half_t)0.f;
    } else if (e < N1 + N2 + N3) {
      int e3 = e - (N1 + N2);
      int r = e3 / K3P, c = e3 % K3P;
      float v = 0.f;
      if (r < 144) {                       // ktd base: only xc_pose cols
        if (c < 1024) v = ktd_w[(size_t)r * 1078 + c];
      } else if (r < 154) {                // decshape: xc_shape cols
        if (c >= 1024 && c < 2048) v = decshape_w[(size_t)(r - 144) * 1024 + (c - 1024)];
      } else if (r < 157) {                // deccam: all 2051 cols
        if (c < 2051) v = deccam_w[(size_t)(r - 154) * 2051 + c];
      }
      W3[e3] = (half_t)v;
    } else {
      int r = e - (N1 + N2 + N3);
      float v = 0.f;
      if (r < 144) v = ktd_b[r];
      else if (r < 154) v = decshape_b[r - 144];
      else if (r < 157) v = deccam_b[r - 154];
      bias3[r] = v;
    }
  }
}

// ---------------- WMMA GEMM: D = A[M,K] * W[N,K]^T + bias ----------------
// 128x64 block tile, 256 threads (8 waves as 4x2, each 2x2 C-tiles), BK=64.
// Tiles staged by TDM (tensor_load_to_lds), double-buffered, 1 barrier/iter.
__global__ __launch_bounds__(256)
void wmma_gemm(const half_t* __restrict__ A, int lda,
               const half_t* __restrict__ W, int ldw,
               const float* __restrict__ bias,
               float* __restrict__ outF, int ldoF,
               half_t* __restrict__ outH, int ldoH,
               int N, int K) {
  __shared__ half_t lds[2][(128 + 64) * LDSP];
  const int t = threadIdx.x;
  const int lane = t & 31;
  const int wv = t >> 5;
  const int rowBase = blockIdx.x * 128;
  const int colBase = blockIdx.y * 64;
  const int tmr = (wv & 3) * 32;    // A-row range of this wave (2 tiles of 16)
  const int tnc = (wv >> 2) * 32;   // B-col range of this wave (2 tiles of 16)
  const int mrow = lane & 15;
  const int khi = (lane >> 4) << 3; // +8 for upper half-wave (K and M offsets)

  v8f acc[2][2] = {};

  const half_t* gA = A + (size_t)rowBase * lda;
  const half_t* gB = W + (size_t)colBase * ldw;
  unsigned offA[2], offB[2];
  offA[0] = (unsigned)(size_t)&lds[0][0];
  offA[1] = (unsigned)(size_t)&lds[1][0];
  offB[0] = offA[0] + 128 * LDSP * sizeof(half_t);
  offB[1] = offA[1] + 128 * LDSP * sizeof(half_t);

  const int niter = K / BK;
  if (wv == 0) {                                    // preload k0 = 0 -> buf 0
    tdm_load_tile64(gA, offA[0], 128, lda);
    tdm_load_tile64(gB, offB[0], 64, ldw);
  }

  for (int i = 0; i < niter; ++i) {
    const int b = i & 1;
    if (wv == 0) __builtin_amdgcn_s_wait_tensorcnt((short)0);
    __syncthreads();                                // publish buf b
    if (wv == 0 && i + 1 < niter) {                 // prefetch next K tile
      tdm_load_tile64(gA + (size_t)(i + 1) * BK, offA[(i + 1) & 1], 128, lda);
      tdm_load_tile64(gB + (size_t)(i + 1) * BK, offB[(i + 1) & 1], 64, ldw);
    }
    const half_t* bufA = &lds[b][0];
    const half_t* bufB = &lds[b][128 * LDSP];
#pragma unroll
    for (int s = 0; s < 2; ++s) {                   // two K=32 steps per buffer
      const int ks = s * 32;
      FragH fa[2], fb[2];
#pragma unroll
      for (int p = 0; p < 8; ++p) {
        int k = ks + (p << 1) + ((p & 4) ? 8 : 0) + khi;
        fa[0].u[p] = *(const unsigned*)(bufA + (tmr + mrow) * LDSP + k);
        fa[1].u[p] = *(const unsigned*)(bufA + (tmr + 16 + mrow) * LDSP + k);
        fb[0].u[p] = *(const unsigned*)(bufB + (tnc + mrow) * LDSP + k);
        fb[1].u[p] = *(const unsigned*)(bufB + (tnc + 16 + mrow) * LDSP + k);
      }
#pragma unroll
      for (int r = 0; r < 2; ++r)
#pragma unroll
        for (int c = 0; c < 2; ++c)
          acc[r][c] = __builtin_amdgcn_wmma_f32_16x16x32_f16(
              false, fa[r].v, false, fb[c].v, (short)0, acc[r][c], false, false);
    }
  }

#pragma unroll
  for (int r = 0; r < 2; ++r) {
#pragma unroll
    for (int c = 0; c < 2; ++c) {
      int col = colBase + tnc + c * 16 + mrow;
      if (col < N) {
        float bval = bias[col];
#pragma unroll
        for (int i = 0; i < 8; ++i) {
          int row = rowBase + tmr + r * 16 + i + khi; // C: VGPR i -> M=i (+8 upper)
          float v = acc[r][c][i] + bval;
          if (outF) outF[(size_t)row * ldoF + col] = v;
          if (outH) outH[(size_t)row * ldoH + col] = (half_t)v;
        }
      }
    }
  }
}

// ---------------- sequential kinematic-tree chain + output assembly ----------
#define NBLK 119            // sum over joints of (n_ancestors + 1)
#define CWSZ (NBLK * 36)    // 4284 floats of 6x6 chain weights
__global__ __launch_bounds__(256)
void ktd_chain(const float* __restrict__ out3,
               const float* __restrict__ init_pose,
               const float* __restrict__ init_shape,
               const float* __restrict__ init_cam,
               const float* __restrict__ ktd_w,
               float* __restrict__ out, int M) {
  __shared__ float cw[CWSZ];
  constexpr int BLK_OFF[25] = {0,1,3,5,7,10,13,16,20,24,28,33,38,43,48,
                               53,59,65,71,78,85,93,101,110,119};
  for (int e = threadIdx.x; e < CWSZ; e += 256) {
    int blk = e / 36, r = e % 36;
    int j = 0;
    while (blk >= BLK_OFF[j + 1]) ++j;
    int b = blk - BLK_OFF[j];
    int o = r / 6, i = r % 6;
    cw[e] = ktd_w[(size_t)(j * 6 + o) * 1078 + 1024 + 6 * b + i];
  }
  __syncthreads();

  int row = blockIdx.x * 256 + threadIdx.x;
  if (row >= M) return;

  constexpr int NANC[24] = {0,1,1,1,2,2,2,3,3,3,4,4,4,4,4,5,5,5,6,6,7,7,8,8};
  constexpr int ANC[24][8] = {
    {0,0,0,0,0,0,0,0},{0,0,0,0,0,0,0,0},{0,0,0,0,0,0,0,0},{0,0,0,0,0,0,0,0},
    {0,1,0,0,0,0,0,0},{0,2,0,0,0,0,0,0},{0,3,0,0,0,0,0,0},
    {0,1,4,0,0,0,0,0},{0,2,5,0,0,0,0,0},{0,3,6,0,0,0,0,0},
    {0,1,4,7,0,0,0,0},{0,2,5,8,0,0,0,0},{0,3,6,9,0,0,0,0},
    {0,3,6,9,0,0,0,0},{0,3,6,9,0,0,0,0},
    {0,3,6,9,12,0,0,0},{0,3,6,9,13,0,0,0},{0,3,6,9,14,0,0,0},
    {0,3,6,9,13,16,0,0},{0,3,6,9,14,17,0,0},
    {0,3,6,9,13,16,18,0},{0,3,6,9,14,17,19,0},
    {0,3,6,9,13,16,18,20},{0,3,6,9,14,17,19,21}};

  const float* o3  = out3 + (size_t)row * LD3;
  const float* ipr = init_pose + (size_t)row * 144;

  float pose[144];
#pragma unroll
  for (int j = 0; j < 24; ++j) {
    float acc[6];
#pragma unroll
    for (int o = 0; o < 6; ++o) acc[o] = o3[j * 6 + o]; // base GEMM incl. bias
    const int boff = BLK_OFF[j] * 36;
#pragma unroll
    for (int p = 0; p < NANC[j]; ++p) {
      const int a = ANC[j][p];
#pragma unroll
      for (int i = 0; i < 6; ++i) {
        float pv = pose[a * 6 + i];
#pragma unroll
        for (int o = 0; o < 6; ++o)
          acc[o] += cw[boff + p * 36 + o * 6 + i] * pv;
      }
    }
#pragma unroll
    for (int i = 0; i < 6; ++i) {
      float pv = ipr[j * 6 + i];
#pragma unroll
      for (int o = 0; o < 6; ++o)
        acc[o] += cw[boff + NANC[j] * 36 + o * 6 + i] * pv;
    }
#pragma unroll
    for (int o = 0; o < 6; ++o) pose[j * 6 + o] = acc[o];
  }

  float* orow = out + (size_t)row * 157;
#pragma unroll
  for (int c = 0; c < 3; ++c) orow[c] = o3[154 + c] + init_cam[(size_t)row * 3 + c];
#pragma unroll
  for (int k = 0; k < 144; ++k) orow[3 + k] = pose[k] + ipr[k];
#pragma unroll
  for (int o = 0; o < 10; ++o) orow[147 + o] = o3[144 + o] + init_shape[(size_t)row * 10 + o];
}

// ---------------- host launch ----------------
extern "C" void kernel_launch(void* const* d_in, const int* in_sizes, int n_in,
                              void* d_out, int out_size, void* d_ws, size_t ws_size,
                              hipStream_t stream) {
  (void)n_in; (void)out_size; (void)ws_size;
  const float* x          = (const float*)d_in[0];
  const float* init_pose  = (const float*)d_in[1];
  const float* init_shape = (const float*)d_in[2];
  const float* init_cam   = (const float*)d_in[3];
  const float* fc1_w      = (const float*)d_in[4];
  const float* fc1_b      = (const float*)d_in[5];
  const float* fc2_w      = (const float*)d_in[6];
  const float* fc2_b      = (const float*)d_in[7];
  const float* decshape_w = (const float*)d_in[8];
  const float* decshape_b = (const float*)d_in[9];
  const float* deccam_w   = (const float*)d_in[10];
  const float* deccam_b   = (const float*)d_in[11];
  const float* ktd_w      = (const float*)d_in[12];
  const float* ktd_b      = (const float*)d_in[13];
  float* out = (float*)d_out;

  const int M = in_sizes[0] / 256; // 32768 (multiple of 128)

  char* ws = (char*)d_ws;
  size_t off = 0;
  auto alloc = [&](size_t bytes) {
    size_t p = off;
    off = (off + bytes + 255) & ~(size_t)255;
    return p;
  };
  half_t* A1    = (half_t*)(ws + alloc((size_t)M * K1P * 2));
  half_t* A2    = (half_t*)(ws + alloc((size_t)M * K2P * 2));
  half_t* A3    = (half_t*)(ws + alloc((size_t)M * K3P * 2));
  half_t* W1    = (half_t*)(ws + alloc((size_t)1024 * K1P * 2));
  half_t* W2    = (half_t*)(ws + alloc((size_t)1024 * K2P * 2));
  half_t* W3    = (half_t*)(ws + alloc((size_t)N3P * K3P * 2));
  float*  bias3 = (float*) (ws + alloc((size_t)N3P * 4));
  float*  out3  = (float*) (ws + alloc((size_t)M * LD3 * 4));

  // 1) pack activations (f32 -> f16, concat + pad)
  {
    long total = (long)M * (K1P + K2P + 64);
    int blocks = (int)((total + 255) / 256);
    if (blocks > 65535) blocks = 65535;
    pack_activations<<<blocks, 256, 0, stream>>>(x, init_pose, init_shape, init_cam,
                                                 A1, A2, A3, M);
  }
  // 2) pack weights
  {
    int total = 1024 * K1P + 1024 * K2P + N3P * K3P + N3P;
    int blocks = (total + 255) / 256;
    pack_weights<<<blocks, 256, 0, stream>>>(fc1_w, fc2_w, ktd_w, decshape_w, deccam_w,
                                             ktd_b, decshape_b, deccam_b,
                                             W1, W2, W3, bias3);
  }
  // 3) fc1: xc_shape_cam -> A3 cols [1024,2048) as f16
  wmma_gemm<<<dim3(M / 128, 1024 / 64), 256, 0, stream>>>(
      A1, K1P, W1, K1P, fc1_b, nullptr, 0, A3 + 1024, K3P, 1024, K1P);
  // 4) fc2: xc_pose_cam -> A3 cols [0,1024) as f16
  wmma_gemm<<<dim3(M / 128, 1024 / 64), 256, 0, stream>>>(
      A2, K2P, W2, K2P, fc2_b, nullptr, 0, A3, K3P, 1024, K2P);
  // 5) combined [ktd-base | decshape | deccam] GEMM -> out3 (f32)
  wmma_gemm<<<dim3(M / 128, N3P / 64), 256, 0, stream>>>(
      A3, K3P, W3, K3P, bias3, out3, LD3, nullptr, 0, LD3, K3P);
  // 6) sequential kinematic chain + final output
  ktd_chain<<<(M + 255) / 256, 256, 0, stream>>>(out3, init_pose, init_shape,
                                                 init_cam, ktd_w, out, M);
}
